// Attentionlayer_84576495993011
// MI455X (gfx1250) — matrix operand order repaired and checked by reading, not compiled, not run on hardware
//
#include <hip/hip_runtime.h>
#include <hip/hip_bf16.h>

#define B_  4
#define S_  4096
#define E_  64
#define D_  512
#define MQ_ 32                      // query rows per workgroup
#define NK_ 64                      // keys per outer iteration (4 slices x 16)
#define NKBLK_ (S_ / NK_)           // 64

typedef __attribute__((ext_vector_type(16))) _Float16 v16h;
typedef __attribute__((ext_vector_type(8)))  _Float16 v8h;
typedef __attribute__((ext_vector_type(8)))  float    v8f;
typedef __attribute__((ext_vector_type(4)))  int      v4i;
typedef __attribute__((address_space(1))) v4i* gas_v4i_p;   // global (__device__)
typedef __attribute__((address_space(3))) v4i* las_v4i_p;   // LDS (__shared__)

__device__ __forceinline__ v8f wmma_f16(v16h a, v16h b, v8f c) {
  return __builtin_amdgcn_wmma_f32_16x16x32_f16(false, a, false, b, (short)0, c,
                                                false, false);
}

// A fragment 16x32 f16: lane holds row m=lane&15.
// element e -> K = e + (e>=8 ? 8 : 0) + (lane>=16 ? 8 : 0)   (two 8-half chunks)
// p points at &A[m][k0]; ahi = (lane&16)?8:0.
__device__ __forceinline__ v16h load_a16(const _Float16* p, int ahi) {
  v8h lo = *(const v8h*)(p + ahi);
  v8h h2 = *(const v8h*)(p + ahi + 16);
  v16h a;
#pragma unroll
  for (int e = 0; e < 8; ++e) { a[e] = lo[e]; a[e + 8] = h2[e]; }
  return a;
}

// B fragment 32x16 f16 from COLUMN-MAJOR storage (Bt[n][k]):
// lane holds column n=lane&15; element e -> K = e + (e>=8 ? 8 : 0) + (lane>=16 ? 8 : 0)
// p points at &Bt[n][k0]; bhi = (lane&16)?16:0.
__device__ __forceinline__ v16h load_b16(const _Float16* p, int bhi) {
  v8h lo = *(const v8h*)(p + (bhi >> 1));
  v8h h2 = *(const v8h*)(p + (bhi >> 1) + 16);
  v16h b;
#pragma unroll
  for (int e = 0; e < 8; ++e) { b[e] = lo[e]; b[e + 8] = h2[e]; }
  return b;
}

// ---------------- stage 1: conversions ----------------

__global__ void cvt_f32_to_f16(const float* __restrict__ in,
                               _Float16* __restrict__ out, int n) {
  int i = blockIdx.x * blockDim.x + threadIdx.x;
  if (i < n) out[i] = (_Float16)in[i];
}

// in: [R][C] f32 row-major  ->  out: [C][R] f16 (column-major of in)
__global__ void cvt_transpose(const float* __restrict__ in,
                              _Float16* __restrict__ out, int R, int C) {
  int i = blockIdx.x * blockDim.x + threadIdx.x;
  if (i < R * C) {
    int r = i / C, c = i % C;
    out[(size_t)c * R + r] = (_Float16)in[i];
  }
}

// ---------------- stage 2: q/k/v projection ----------------

__global__ __launch_bounds__(256)
void proj_qkv(const _Float16* __restrict__ xh,
              const _Float16* __restrict__ WqT, const _Float16* __restrict__ WkT,
              const _Float16* __restrict__ WvT,
              const float* __restrict__ bq, const float* __restrict__ bk,
              const float* __restrict__ bv,
              _Float16* __restrict__ qh, _Float16* __restrict__ kh,
              _Float16* __restrict__ vT) {
  const int TM = (B_ * S_) / 16;      // 1024
  const int TN = D_ / 16;             // 32
  int gw = blockIdx.x * (blockDim.x / 32) + (threadIdx.x >> 5);
  if (gw >= 3 * TM * TN) return;
  const int lane = threadIdx.x & 31;
  const int ahi = (lane & 16) ? 8 : 0;
  const int bhi = (lane & 16) ? 16 : 0;
  const int mat  = gw / (TM * TN);
  const int rest = gw % (TM * TN);
  const int tm = rest / TN, tn = rest % TN;
  const int rowbase = tm * 16, colbase = tn * 16;

  const _Float16* WT  = (mat == 0) ? WqT : (mat == 1) ? WkT : WvT;
  const float*    bia = (mat == 0) ? bq  : (mat == 1) ? bk  : bv;

  const _Float16* arow = xh + (size_t)(rowbase + (lane & 15)) * E_;
  const _Float16* brow = WT + (size_t)(colbase + (lane & 15)) * E_;  // WT: [D][E]

  v8f c = {};
  c = wmma_f16(load_a16(arow,      ahi), load_b16(brow,      bhi), c);
  c = wmma_f16(load_a16(arow + 32, ahi), load_b16(brow + 32, bhi), c);

  const float bn = bia[colbase + (lane & 15)];
  if (mat < 2) {
    _Float16* out = (mat == 0) ? qh : kh;
#pragma unroll
    for (int r = 0; r < 8; ++r)
      out[(size_t)(rowbase + r + ahi) * D_ + colbase + (lane & 15)] =
          (_Float16)(c[r] + bn);
  } else {
    int b  = rowbase / S_;
    int s0 = rowbase % S_;
    int d  = colbase + (lane & 15);
    _Float16* col = vT + ((size_t)b * D_ + d) * S_ + s0;
#pragma unroll
    for (int r = 0; r < 8; ++r) col[r + ahi] = (_Float16)(c[r] + bn);
  }
}

// ---------------- stage 3: fused flash attention ----------------
// 512 workgroups (one per 32 query rows), 8 waves each.
// wave w: qg = w>>2 selects 16-row query sub-tile; sl = w&3 is the key slice
// for the score GEMM and the 128-column slice for the p@v GEMM.

__global__ __launch_bounds__(256)
void flash_attn(const _Float16* __restrict__ qh, const _Float16* __restrict__ kh,
                const _Float16* __restrict__ vT, const float* __restrict__ mask,
                _Float16* __restrict__ zh) {
  const int b  = blockIdx.x / (S_ / MQ_);
  const int q0 = (blockIdx.x % (S_ / MQ_)) * MQ_;
  const int tid  = threadIdx.x;
  const int w    = tid >> 5;
  const int lane = tid & 31;
  const int qg   = w >> 2;                 // 0..1
  const int sl   = w & 3;                  // 0..3
  const int ln16 = lane & 15;
  const int ahi  = (lane & 16) ? 8  : 0;   // A K-offset, also C/D row offset
  const int bhi  = (lane & 16) ? 16 : 0;   // B K-offset

  __shared__ _Float16 q_lds[MQ_ * D_];     // 32 KB, resident all kernel
  __shared__ float    sc[MQ_][NK_ + 4];    // raw scores (padded)
  __shared__ _Float16 pl[MQ_][NK_];        // exp-probs f16 (A of p@v)
  __shared__ float    alpha_s[MQ_];
  __shared__ float    scale_s[MQ_];

  // ---- stage the q tile (32x512 f16) into LDS, async path if available
  const _Float16* gq = qh + (size_t)(b * S_ + q0) * D_;
#if __has_builtin(__builtin_amdgcn_global_load_async_to_lds_b128)
  for (int i = tid; i < (MQ_ * D_) / 8; i += 256) {
    __builtin_amdgcn_global_load_async_to_lds_b128(
        (gas_v4i_p)(gq + i * 8), (las_v4i_p)(q_lds + i * 8), 0, 0);
  }
#if __has_builtin(__builtin_amdgcn_s_wait_asynccnt)
  __builtin_amdgcn_s_wait_asynccnt(0);
#else
  asm volatile("s_wait_asynccnt 0" ::: "memory");
#endif
#else
  for (int i = tid; i < (MQ_ * D_) / 8; i += 256)
    *(v8h*)(q_lds + i * 8) = *(const v8h*)(gq + i * 8);
#endif
  __syncthreads();

  const int srow = tid >> 3;               // softmax: 8 threads per row
  const int ssub = tid & 7;
  float mrow = -3.0e38f, lrow = 0.0f;      // running max / sum for row `srow`
  v8f acc[8] = {};                         // 16 rows x 128 cols per wave

  for (int kb = 0; kb < NKBLK_; ++kb) {
    // ---- scores: 16x16 tile = q(qg) . k(slice)^T, two independent chains
    const int key0 = kb * NK_ + sl * 16;
    const _Float16* krow = kh + ((size_t)(b * S_ + key0) + ln16) * D_;
    __builtin_prefetch(krow + (size_t)NK_ * D_, 0, 0);   // next key block
    const _Float16* qrow = q_lds + (size_t)(qg * 16 + ln16) * D_;
    v8f s0 = {}, s1 = {};
#pragma unroll
    for (int kk = 0; kk < 16; kk += 2) {
      s0 = wmma_f16(load_a16(qrow + kk * 32, ahi),
                    load_b16(krow + kk * 32, bhi), s0);
      s1 = wmma_f16(load_a16(qrow + (kk + 1) * 32, ahi),
                    load_b16(krow + (kk + 1) * 32, bhi), s1);
    }
    const float* mp = mask + (size_t)(b * S_ + q0 + qg * 16) * S_ + key0 + ln16;
#pragma unroll
    for (int r = 0; r < 8; ++r)
      sc[qg * 16 + r + ahi][sl * 16 + ln16] = s0[r] + s1[r] + mp[(size_t)(r + ahi) * S_];
    __syncthreads();

    // ---- online softmax, all 256 threads (8 per row, shuffle reductions)
    {
      const int c0 = ssub * 8;
      float lm = -3.0e38f;
#pragma unroll
      for (int j = 0; j < 8; ++j) lm = fmaxf(lm, sc[srow][c0 + j]);
      lm = fmaxf(lm, __shfl_xor(lm, 1));
      lm = fmaxf(lm, __shfl_xor(lm, 2));
      lm = fmaxf(lm, __shfl_xor(lm, 4));
      float nm = fmaxf(mrow, lm);
      float al = __expf(mrow - nm);
      float sum = 0.0f;
#pragma unroll
      for (int j = 0; j < 8; ++j) {
        float p = __expf(sc[srow][c0 + j] - nm);
        pl[srow][c0 + j] = (_Float16)p;
        sum += p;
      }
      sum += __shfl_xor(sum, 1);
      sum += __shfl_xor(sum, 2);
      sum += __shfl_xor(sum, 4);
      lrow = lrow * al + sum;
      mrow = nm;
      if (ssub == 0) alpha_s[srow] = al;
    }
    __syncthreads();

    // ---- rescale accumulators, then acc += p(16xNK) . v(NK x 128-slice)
    float alv[8];
#pragma unroll
    for (int r = 0; r < 8; ++r) alv[r] = alpha_s[qg * 16 + r + ahi];
#pragma unroll
    for (int t = 0; t < 8; ++t)
#pragma unroll
      for (int r = 0; r < 8; ++r) acc[t][r] *= alv[r];

    const int cb = sl * 128;
#pragma unroll
    for (int t = 0; t < 8; ++t) {
      const int col = cb + t * 16 + ln16;
      const _Float16* vcol = vT + ((size_t)b * D_ + col) * S_ + kb * NK_;
      const _Float16* prow = &pl[qg * 16 + ln16][0];
#pragma unroll
      for (int kk = 0; kk < 2; ++kk)
        acc[t] = wmma_f16(load_a16(prow + kk * 32, ahi),
                          load_b16(vcol + kk * 32, bhi), acc[t]);
    }
    __syncthreads();   // protect sc/pl before next iteration overwrites
  }

  // ---- normalize by row-sum and the reference's 1/E^0.25, emit f16 z
  if (ssub == 0) scale_s[srow] = 0.35355339059327373f / lrow;
  __syncthreads();
  float sv[8];
#pragma unroll
  for (int r = 0; r < 8; ++r) sv[r] = scale_s[qg * 16 + r + ahi];
  const int cb = sl * 128;
#pragma unroll
  for (int t = 0; t < 8; ++t) {
    const int col = cb + t * 16 + ln16;
#pragma unroll
    for (int r = 0; r < 8; ++r)
      zh[(size_t)(b * S_ + q0 + qg * 16 + r + ahi) * D_ + col] =
          (_Float16)(acc[t][r] * sv[r]);
  }
}

// ---------------- stage 4: output projection ----------------

__global__ __launch_bounds__(256)
void proj_out(const _Float16* __restrict__ zh, const _Float16* __restrict__ WoT,
              const float* __restrict__ bo, float* __restrict__ out) {
  const int TM = (B_ * S_) / 16;   // 1024
  const int TN = E_ / 16;          // 4
  int gw = blockIdx.x * (blockDim.x / 32) + (threadIdx.x >> 5);
  if (gw >= TM * TN) return;
  const int lane = threadIdx.x & 31;
  const int ahi = (lane & 16) ? 8 : 0;
  const int bhi = (lane & 16) ? 16 : 0;
  const int tm = gw / TN, tn = gw % TN;

  const _Float16* arow = zh  + (size_t)(tm * 16 + (lane & 15)) * D_;
  const _Float16* brow = WoT + (size_t)(tn * 16 + (lane & 15)) * D_;  // [E][D]

  v8f c0 = {}, c1 = {};
#pragma unroll
  for (int kk = 0; kk < 16; kk += 2) {
    c0 = wmma_f16(load_a16(arow + kk * 32, ahi),
                  load_b16(brow + kk * 32, bhi), c0);
    c1 = wmma_f16(load_a16(arow + (kk + 1) * 32, ahi),
                  load_b16(brow + (kk + 1) * 32, bhi), c1);
  }
  const float bn = bo[tn * 16 + (lane & 15)];
#pragma unroll
  for (int r = 0; r < 8; ++r)
    out[(size_t)(tm * 16 + r + ahi) * E_ + tn * 16 + (lane & 15)] =
        c0[r] + c1[r] + bn;
}

// ---------------- host launcher ----------------

extern "C" void kernel_launch(void* const* d_in, const int* in_sizes, int n_in,
                              void* d_out, int out_size, void* d_ws, size_t ws_size,
                              hipStream_t stream) {
  const float* x    = (const float*)d_in[0];
  const float* mask = (const float*)d_in[1];
  const float* Wq   = (const float*)d_in[2];
  const float* bq   = (const float*)d_in[3];
  const float* Wk   = (const float*)d_in[4];
  const float* bk   = (const float*)d_in[5];
  const float* Wv   = (const float*)d_in[6];
  const float* bv   = (const float*)d_in[7];
  const float* Wo   = (const float*)d_in[8];
  const float* bo   = (const float*)d_in[9];
  float* out = (float*)d_out;

  char* ws = (char*)d_ws;
  const size_t nX  = (size_t)B_ * S_ * E_;       // 1,048,576
  const size_t nW  = (size_t)E_ * D_;            // 32,768
  const size_t nQK = (size_t)B_ * S_ * D_;       // 8,388,608
  _Float16* xh  = (_Float16*)(ws);
  _Float16* WqT = (_Float16*)(ws + 2 * nX);
  _Float16* WkT = (_Float16*)(ws + 2 * nX + 2 * nW);
  _Float16* WvT = (_Float16*)(ws + 2 * nX + 4 * nW);
  _Float16* WoT = (_Float16*)(ws + 2 * nX + 6 * nW);
  char* big = ws + 2 * nX + 8 * nW;
  _Float16* qh = (_Float16*)(big);
  _Float16* kh = (_Float16*)(big + 2 * nQK);
  _Float16* vT = (_Float16*)(big + 4 * nQK);
  _Float16* zh = (_Float16*)(big + 6 * nQK);

  cvt_f32_to_f16<<<(int)((nX + 255) / 256), 256, 0, stream>>>(x, xh, (int)nX);
  cvt_transpose<<<(int)((nW + 255) / 256), 256, 0, stream>>>(Wq, WqT, E_, D_);
  cvt_transpose<<<(int)((nW + 255) / 256), 256, 0, stream>>>(Wk, WkT, E_, D_);
  cvt_transpose<<<(int)((nW + 255) / 256), 256, 0, stream>>>(Wv, WvT, E_, D_);
  cvt_transpose<<<(int)((nW + 255) / 256), 256, 0, stream>>>(Wo, WoT, D_, E_);

  {
    int waves = 3 * ((B_ * S_) / 16) * (D_ / 16);   // 98304
    proj_qkv<<<waves / 8, 256, 0, stream>>>(xh, WqT, WkT, WvT, bq, bk, bv,
                                            qh, kh, vT);
  }
  flash_attn<<<(B_ * S_) / MQ_, 256, 0, stream>>>(qh, kh, vT, mask, zh);
  {
    int waves = ((B_ * S_) / 16) * (E_ / 16);       // 4096
    proj_out<<<waves / 8, 256, 0, stream>>>(zh, WoT, bo, out);
  }
}
